// CausalAttentionWithDropout_55894704390358
// MI455X (gfx1250) — matrix operand order repaired and checked by reading, not compiled, not run on hardware
//
#include <hip/hip_runtime.h>
#include <hip/hip_bf16.h>
#include <math.h>

// ---------------------------------------------------------------------------
// CausalAttentionWithDropout on MI455X (gfx1250, wave32, WMMA bf16 path)
//   out = softmax_causal((x Wq^T)(x Wk^T)^T / sqrt(D)) * drop_mask @ (x Wv^T)
// GEMM staging: Tensor Data Mover (tensor_load_to_lds) with LDS row padding.
// ---------------------------------------------------------------------------

#define USE_TDM 1   // 1: TDM tensor_load_to_lds staging; 0: per-lane async b128

typedef __bf16 bf16_t;
typedef __bf16  bf16x16 __attribute__((ext_vector_type(16)));
typedef float   f32x8   __attribute__((ext_vector_type(8)));
typedef unsigned int u32x4 __attribute__((ext_vector_type(4)));
typedef unsigned int u32x8 __attribute__((ext_vector_type(8)));

constexpr int BM = 128, BN = 128, BK = 32, LDP = 40; // LDP: padded LDS row pitch (elems)

// ---- per-lane async 16B global -> LDS copy (fallback path, ASYNCcnt) ------
__device__ __forceinline__ void cp16_g2l(void* lds, const void* g) {
  unsigned int l = (unsigned int)(unsigned long long)(uintptr_t)lds; // LDS byte offset
  unsigned long long ga = (unsigned long long)(uintptr_t)g;
  asm volatile("global_load_async_to_lds_b128 %0, %1, off"
               :: "v"(l), "v"(ga) : "memory");
}

// ---- TDM: one descriptor moves a 128x32 bf16 tile into LDS (TENSORcnt) ----
// D# per CDNA5 ISA. Pad every 64B row by 16B so
// the LDS pitch is 80B (= LDP elements), matching the WMMA fragment layout.
__device__ __forceinline__ void tdm_load_2d(void* ldsBase, const bf16_t* gTile,
                                            unsigned rowLenElems,   // tensor_dim0 & stride0 (= K)
                                            unsigned nRowsTensor) { // tensor_dim1
  unsigned long long ga = (unsigned long long)(uintptr_t)gTile;
  unsigned lds = (unsigned)(uintptr_t)ldsBase;
  u32x4 g0;
  g0[0] = 1u;                                                 // count=1 (valid user D#)
  g0[1] = lds;                                                // lds_addr
  g0[2] = (unsigned)ga;                                       // global_addr[31:0]
  g0[3] = ((unsigned)(ga >> 32) & 0x01ffffffu) | (2u << 30);  // addr[56:32] | type=2
  u32x8 g1;
  g1[0] = (1u << 16)            // data_size = 1 -> 2 bytes
        | (1u << 20)            // pad_enable
        | (3u << 22)            // pad_interval code 3 -> every 16 DWORDs (64B row)
        | (3u << 25);           // pad_amount  code 3 -> 4 DWORDs (16B pad)
  g1[1] = (rowLenElems & 0xffffu) << 16;                          // tensor_dim0[15:0]
  g1[2] = (rowLenElems >> 16) | ((nRowsTensor & 0xffffu) << 16);  // tdim0 hi | tdim1 lo
  g1[3] = (nRowsTensor >> 16) | ((unsigned)BK << 16);             // tdim1 hi | tile_dim0=32
  g1[4] = (unsigned)BM;                                           // tile_dim1=128, tile_dim2=0
  g1[5] = rowLenElems;                                            // tensor_dim0_stride[31:0]
  g1[6] = 0u;                                                     // stride hi / dim1_stride lo
  g1[7] = 0u;                                                     // dim1_stride hi
  asm volatile("tensor_load_to_lds %0, %1" :: "s"(g0), "s"(g1) : "memory");
}

// ---------------------------------------------------------------------------
// GEMM-NT: C[M,N] = scale * (A[M,K] @ B[N,K]^T), bf16 inputs, f32 accum.
// Block tile 128x128, K-step 32, 256 threads = 8 waves in a 2(M)x4(N) grid.
// Each wave: 4 M-tiles x 2 N-tiles of v_wmma_f32_16x16x32_bf16.
// ---------------------------------------------------------------------------
template <typename CT, bool CAUSAL_SKIP, bool KLIMIT_BY_ROW>
__global__ __launch_bounds__(256)
void gemm_nt_kernel(const bf16_t* __restrict__ A, const bf16_t* __restrict__ B,
                    CT* __restrict__ C, int M, int N, int K, float scale) {
  const int m0 = blockIdx.y * BM;
  const int n0 = blockIdx.x * BN;
  if (CAUSAL_SKIP && n0 > m0 + (BM - 1)) return;  // whole block strictly above diag

  __shared__ __align__(16) bf16_t lA[2][BM * LDP];
  __shared__ __align__(16) bf16_t lB[2][BN * LDP];

  const int tid  = threadIdx.x;
  const int lane = tid & 31;
  const int wave = tid >> 5;
  const int wm   = wave & 1;   // 64-row strip
  const int wn   = wave >> 1;  // 32-col strip

  int kEnd = K;
  if (KLIMIT_BY_ROW) { int lim = m0 + BM; kEnd = lim < K ? lim : K; }
  const int nk = kEnd / BK;

  f32x8 acc[4][2] = {};

#if USE_TDM
  // TDM staging: wave 0 pulls the A tile, wave 1 the B tile (scalar-gated:
  // TDM ignores EXEC, so the branch condition must be wave-uniform).
  const int wid = __builtin_amdgcn_readfirstlane(tid >> 5);
  auto stage = [&](int buf, int k0) {
    if (wid == 0)
      tdm_load_2d(&lA[buf][0], A + (size_t)m0 * K + k0, (unsigned)K, (unsigned)M);
    else if (wid == 1)
      tdm_load_2d(&lB[buf][0], B + (size_t)n0 * K + k0, (unsigned)K, (unsigned)N);
  };
  auto stage_wait = [&]() {
    if (wid < 2) __builtin_amdgcn_s_wait_tensorcnt(0);
  };
#else
  // per-lane async staging: each thread moves one 32B chunk of A and of B
  const int crow = tid >> 1;
  const int ckc  = (tid & 1) * 16;
  const bf16_t* gA = A + (size_t)(m0 + crow) * K + ckc;
  const bf16_t* gB = B + (size_t)(n0 + crow) * K + ckc;
  auto stage = [&](int buf, int k0) {
    const bf16_t* ga = gA + k0;
    const bf16_t* gb = gB + k0;
    bf16_t* sa = &lA[buf][crow * LDP + ckc];
    bf16_t* sb = &lB[buf][crow * LDP + ckc];
    cp16_g2l(sa,     ga);
    cp16_g2l(sa + 8, ga + 8);
    cp16_g2l(sb,     gb);
    cp16_g2l(sb + 8, gb + 8);
  };
  auto stage_wait = [&]() {
    asm volatile("s_wait_asynccnt 0" ::: "memory");
  };
#endif

  stage(0, 0);
  stage_wait();
  __syncthreads();

  // fragment addressing per CDNA5 16-bit WMMA VGPR layouts
  const int bn = lane & 15, bk = (lane >> 4) * 8;   // B: lane=N, halves hold K 0-7/8-15 (+16)
  const int am = lane & 15, ak = (lane >> 4) * 8;   // A: lane=M, halves hold K 0-7/8-15 (+16)

  for (int kt = 0; kt < nk; ++kt) {
    const int buf = kt & 1;
    if (kt + 1 < nk) stage(buf ^ 1, (kt + 1) * BK);

    union Frag { bf16x16 v; u32x4 q[2]; };
    Frag fb[2], fa[4];
    #pragma unroll
    for (int ni = 0; ni < 2; ++ni) {
      const bf16_t* p = &lB[buf][(wn * 32 + ni * 16 + bn) * LDP + bk];
      fb[ni].q[0] = *(const u32x4*)p;
      fb[ni].q[1] = *(const u32x4*)(p + 16);
    }
    #pragma unroll
    for (int mi = 0; mi < 4; ++mi) {
      const bf16_t* p = &lA[buf][(wm * 64 + mi * 16 + am) * LDP + ak];
      fa[mi].q[0] = *(const u32x4*)p;        // K = ak..ak+7
      fa[mi].q[1] = *(const u32x4*)(p + 16); // K = 16+ak..16+ak+7
    }
    #pragma unroll
    for (int mi = 0; mi < 4; ++mi)
      #pragma unroll
      for (int ni = 0; ni < 2; ++ni)
        acc[mi][ni] = __builtin_amdgcn_wmma_f32_16x16x32_bf16(
            false, fa[mi].v, false, fb[ni].v, (short)0, acc[mi][ni], false, false);

    if (kt + 1 < nk) stage_wait();
    __syncthreads();
  }

  // C/D layout: lane = N (mod 16), VGPR r = row r (lanes 0-15) / 8+r (lanes 16-31)
  const int cn = lane & 15;
  const int cr = (lane >> 4) * 8;
  #pragma unroll
  for (int mi = 0; mi < 4; ++mi) {
    #pragma unroll
    for (int ni = 0; ni < 2; ++ni) {
      const int col   = n0 + wn * 32 + ni * 16 + cn;
      const int rbase = m0 + wm * 64 + mi * 16 + cr;
      #pragma unroll
      for (int r = 0; r < 8; ++r) {
        float v = acc[mi][ni][r] * scale;
        C[(size_t)(rbase + r) * N + col] = (CT)v;
      }
    }
  }
}

// ---------------------------------------------------------------------------
// fp32 -> bf16 convert
// ---------------------------------------------------------------------------
__global__ void f32_to_bf16_kernel(const float* __restrict__ in,
                                   bf16_t* __restrict__ out, size_t n) {
  size_t i = blockIdx.x * (size_t)blockDim.x + threadIdx.x;
  size_t stride = gridDim.x * (size_t)blockDim.x;
  for (; i < n; i += stride) out[i] = (bf16_t)in[i];
}

// ---------------------------------------------------------------------------
// bf16 transpose (32x32 tiles) : out[c*rows + r] = in[r*cols + c]
// ---------------------------------------------------------------------------
__global__ __launch_bounds__(256)
void transpose_bf16_kernel(const bf16_t* __restrict__ in, bf16_t* __restrict__ out,
                           int rows, int cols) {
  __shared__ bf16_t tile[32][33];
  int x = blockIdx.x * 32 + threadIdx.x;
  int y = blockIdx.y * 32 + threadIdx.y;
  #pragma unroll
  for (int j = 0; j < 32; j += 8)
    tile[threadIdx.y + j][threadIdx.x] = in[(size_t)(y + j) * cols + x];
  __syncthreads();
  x = blockIdx.y * 32 + threadIdx.x;
  y = blockIdx.x * 32 + threadIdx.y;
  #pragma unroll
  for (int j = 0; j < 32; j += 8)
    out[(size_t)(y + j) * rows + x] = tile[threadIdx.x][threadIdx.y + j];
}

// ---------------------------------------------------------------------------
// row softmax (causal) + dropout mask, fp32 scores -> bf16 probabilities
// one workgroup (256 threads) per row; full row (4096) held in registers
// ---------------------------------------------------------------------------
__global__ __launch_bounds__(256)
void softmax_drop_kernel(const float* __restrict__ Sraw,
                         const float* __restrict__ drop,
                         bf16_t* __restrict__ P, int S) {
  const int row  = blockIdx.x;
  const int tid  = threadIdx.x;
  const int lane = tid & 31;
  const int wave = tid >> 5;
  const float NEG_INF = -__builtin_inff();

  const float* srow = Sraw + (size_t)row * S;
  float v[16];
  #pragma unroll
  for (int u = 0; u < 16; ++u) {
    int j = tid + u * 256;
    v[u] = (j <= row) ? srow[j] : NEG_INF;
  }

  __shared__ float red[8];

  // row max
  float m = v[0];
  #pragma unroll
  for (int u = 1; u < 16; ++u) m = fmaxf(m, v[u]);
  #pragma unroll
  for (int off = 16; off > 0; off >>= 1) m = fmaxf(m, __shfl_xor(m, off, 32));
  if (lane == 0) red[wave] = m;
  __syncthreads();
  float mAll = red[0];
  #pragma unroll
  for (int w = 1; w < 8; ++w) mAll = fmaxf(mAll, red[w]);
  __syncthreads();

  // exp + row sum
  float ssum = 0.0f;
  #pragma unroll
  for (int u = 0; u < 16; ++u) { v[u] = __expf(v[u] - mAll); ssum += v[u]; }
  #pragma unroll
  for (int off = 16; off > 0; off >>= 1) ssum += __shfl_xor(ssum, off, 32);
  if (lane == 0) red[wave] = ssum;
  __syncthreads();
  float tot = 0.0f;
  #pragma unroll
  for (int w = 0; w < 8; ++w) tot += red[w];
  const float inv = 1.0f / tot;

  const float* drow = drop + (size_t)row * S;
  bf16_t* prow = P + (size_t)row * S;
  #pragma unroll
  for (int u = 0; u < 16; ++u) {
    int j = tid + u * 256;
    float p = v[u] * inv * drow[j];   // v[u]==0 above the diagonal
    prow[j] = (bf16_t)p;
  }
}

// ---------------------------------------------------------------------------
// host-side orchestration
// ---------------------------------------------------------------------------
extern "C" void kernel_launch(void* const* d_in, const int* in_sizes, int n_in,
                              void* d_out, int out_size, void* d_ws, size_t ws_size,
                              hipStream_t stream) {
  (void)in_sizes; (void)n_in; (void)out_size; (void)ws_size;
  const int S = 4096, DIN = 2048, DOUT = 2048;

  const float* x    = (const float*)d_in[0];
  const float* Wq   = (const float*)d_in[1];
  const float* Wk   = (const float*)d_in[2];
  const float* Wv   = (const float*)d_in[3];
  const float* drop = (const float*)d_in[4];
  float* out = (float*)d_out;

  char* ws = (char*)d_ws;
  size_t off = 0;
  auto carve = [&](size_t bytes) -> char* {
    char* p = ws + off;
    off += (bytes + 255) & ~(size_t)255;
    return p;
  };
  bf16_t* xb   = (bf16_t*)carve((size_t)S * DIN * 2);
  bf16_t* Wqb  = (bf16_t*)carve((size_t)DOUT * DIN * 2);
  bf16_t* Wkb  = (bf16_t*)carve((size_t)DOUT * DIN * 2);
  bf16_t* Wvb  = (bf16_t*)carve((size_t)DOUT * DIN * 2);
  bf16_t* Qb   = (bf16_t*)carve((size_t)S * DOUT * 2);
  bf16_t* Kb   = (bf16_t*)carve((size_t)S * DOUT * 2);
  bf16_t* Vb   = (bf16_t*)carve((size_t)S * DOUT * 2);
  bf16_t* Vtb  = (bf16_t*)carve((size_t)DOUT * S * 2);
  float*  Sraw = (float*)carve((size_t)S * S * 4);
  bf16_t* Pb   = (bf16_t*)carve((size_t)S * S * 2);

  // 1) convert inputs to bf16
  f32_to_bf16_kernel<<<2048, 256, 0, stream>>>(x,  xb,  (size_t)S * DIN);
  f32_to_bf16_kernel<<<1024, 256, 0, stream>>>(Wq, Wqb, (size_t)DOUT * DIN);
  f32_to_bf16_kernel<<<1024, 256, 0, stream>>>(Wk, Wkb, (size_t)DOUT * DIN);
  f32_to_bf16_kernel<<<1024, 256, 0, stream>>>(Wv, Wvb, (size_t)DOUT * DIN);

  // 2) Q, K, V projections (bf16 out)
  dim3 blk(256);
  dim3 gQKV(DOUT / BN, S / BM);
  gemm_nt_kernel<bf16_t, false, false><<<gQKV, blk, 0, stream>>>(xb, Wqb, Qb, S, DOUT, DIN, 1.0f);
  gemm_nt_kernel<bf16_t, false, false><<<gQKV, blk, 0, stream>>>(xb, Wkb, Kb, S, DOUT, DIN, 1.0f);
  gemm_nt_kernel<bf16_t, false, false><<<gQKV, blk, 0, stream>>>(xb, Wvb, Vb, S, DOUT, DIN, 1.0f);

  // 3) V^T for the final GEMM-NT
  transpose_bf16_kernel<<<dim3(DOUT / 32, S / 32), dim3(32, 8), 0, stream>>>(Vb, Vtb, S, DOUT);

  // 4) scaled scores, lower-triangular blocks only (f32 out)
  const float scale = 1.0f / sqrtf((float)DOUT);
  gemm_nt_kernel<float, true, false><<<dim3(S / BN, S / BM), blk, 0, stream>>>(Qb, Kb, Sraw, S, S, DOUT, scale);

  // 5) causal softmax + dropout -> P (bf16, zero above diagonal)
  softmax_drop_kernel<<<S, 256, 0, stream>>>(Sraw, drop, Pb, S);

  // 6) out = P @ V  (= P @ (V^T)^T), K-loop limited by causal row block
  gemm_nt_kernel<float, false, true><<<dim3(DOUT / BN, S / BM), blk, 0, stream>>>(Pb, Vtb, out, S, DOUT, S, 1.0f);
}
